// QuanvolutionGen224_65481071400676
// MI455X (gfx1250) — matrix operand order repaired and checked by reading, not compiled, not run on hardware
//
#include <hip/hip_runtime.h>
#include <hip/hip_bf16.h>

typedef __attribute__((ext_vector_type(16))) _Float16 v16h;
typedef __attribute__((ext_vector_type(8)))  _Float16 v8h;
typedef __attribute__((ext_vector_type(8)))  float    v8f;
typedef __attribute__((ext_vector_type(4)))  int      v4i;
typedef __attribute__((address_space(1))) v4i* as1_v4i_p;
typedef __attribute__((address_space(3))) v4i* as3_v4i_p;

// gfx1250 async global->LDS copy path (ASYNCcnt-tracked), with safe fallback.
#if defined(__HIP_DEVICE_COMPILE__) && defined(__has_builtin)
#if __has_builtin(__builtin_amdgcn_global_load_async_to_lds_b128) && \
    __has_builtin(__builtin_amdgcn_s_wait_asynccnt)
#define ASYNC_LDS 1
#endif
#endif
#ifndef ASYNC_LDS
#define ASYNC_LDS 0
#endif

// ---------------------------------------------------------------------------
// Kernel 1: quadrant means (float4 coalesced, wave-parallel) + 4-qubit
// statevector sim (lane 0, fully unrolled, 16 complex amps in registers)
// ---------------------------------------------------------------------------
__global__ __launch_bounds__(256) void k_qfeat(const float* __restrict__ x,
                                               const float* __restrict__ var_params,
                                               float* __restrict__ qf, int B) {
  int wave = threadIdx.x >> 5;
  int lane = threadIdx.x & 31;
  int s = blockIdx.x * 8 + wave;
  if (s >= B) return;                       // wave-uniform branch
  const float4* img4 = (const float4*)(x + (size_t)s * 784);  // 3136B/sample, 16B aligned
  float s00 = 0.f, s01 = 0.f, s10 = 0.f, s11 = 0.f;
  for (int f4 = lane; f4 < 196; f4 += 32) { // b128 loads; 28 % 4 == 0 -> no row cross
    float4 v = img4[f4];
    int e = f4 * 4;
    int r = e / 28, c0 = e - r * 28;
    bool rb = (r >= 14);
    float vs[4] = { v.x, v.y, v.z, v.w };
#pragma unroll
    for (int i = 0; i < 4; i++) {
      bool cb = ((c0 + i) >= 14);
      s00 += (!rb && !cb) ? vs[i] : 0.f;
      s01 += (!rb &&  cb) ? vs[i] : 0.f;
      s10 += ( rb && !cb) ? vs[i] : 0.f;
      s11 += ( rb &&  cb) ? vs[i] : 0.f;
    }
  }
#pragma unroll
  for (int off = 16; off >= 1; off >>= 1) { // wave32 reduction
    s00 += __shfl_xor(s00, off, 32);
    s01 += __shfl_xor(s01, off, 32);
    s10 += __shfl_xor(s10, off, 32);
    s11 += __shfl_xor(s11, off, 32);
  }
  if (lane == 0) {
    const float inv = 1.f / 196.f;
    float ang[4] = { s00 * inv, s01 * inv, s10 * inv, s11 * inv };
    float cq[4], sq[4];
#pragma unroll
    for (int q = 0; q < 4; q++) {
      float th = 0.5f * ang[q];
      cq[q] = __cosf(th); sq[q] = __sinf(th);
    }
    float re[16], im[16];
#pragma unroll
    for (int i = 0; i < 16; i++) {          // product state after the 4 RYs
      float v = 1.f;
#pragma unroll
      for (int q = 0; q < 4; q++) v *= ((i >> (3 - q)) & 1) ? sq[q] : cq[q];
      re[i] = v; im[i] = 0.f;
    }
#pragma unroll
    for (int q = 0; q < 4; q++) {           // RZ(params[q]) then CNOT(q,(q+1)%4)
      float ph = 0.5f * var_params[q];
      float cp = __cosf(ph), sp = __sinf(ph);
#pragma unroll
      for (int i = 0; i < 16; i++) {
        float sg = ((i >> (3 - q)) & 1) ? sp : -sp;
        float r = re[i], m = im[i];
        re[i] = r * cp - m * sg;
        im[i] = r * sg + m * cp;
      }
      int cbm = 1 << (3 - q);
      int tbm = 1 << (3 - ((q + 1) & 3));
#pragma unroll
      for (int i = 0; i < 16; i++) {
        if ((i & cbm) && !(i & tbm)) {
          int j = i | tbm;
          float t;
          t = re[i]; re[i] = re[j]; re[j] = t;
          t = im[i]; im[i] = im[j]; im[j] = t;
        }
      }
    }
#pragma unroll
    for (int q = 0; q < 4; q++) {
      float z = 0.f;
#pragma unroll
      for (int i = 0; i < 16; i++) {
        float p = re[i] * re[i] + im[i] * im[i];
        z += ((i >> (3 - q)) & 1) ? -p : p;
      }
      qf[s * 4 + q] = z;
    }
  }
}

// ---------------------------------------------------------------------------
// Kernel 2: pack w_fc (392x224 f32) -> fp16, N padded to 400 (25 tiles of 16)
// ---------------------------------------------------------------------------
__global__ __launch_bounds__(256) void k_pack_wfc(const float* __restrict__ w_fc,
                                                  _Float16* __restrict__ wfc16) {
  int idx = blockIdx.x * 256 + threadIdx.x;
  if (idx >= 400 * 224) return;
  int n = idx / 224, k = idx - n * 224;
  wfc16[idx] = (_Float16)((n < 392) ? w_fc[n * 224 + k] : 0.f);
}

// ---------------------------------------------------------------------------
// Kernel 3: latent = relu(qf @ w_lat^T + b_lat), K=4 -> scalar VALU, fp16 out
// ---------------------------------------------------------------------------
__global__ __launch_bounds__(256) void k_latent(const float* __restrict__ qf,
                                                const float* __restrict__ w_lat,
                                                const float* __restrict__ b_lat,
                                                _Float16* __restrict__ lat, int B) {
  int idx = blockIdx.x * 256 + threadIdx.x;
  if (idx >= B * 224) return;
  int s = idx / 224, j = idx - s * 224;
  const float* q = qf + (size_t)s * 4;
  const float* w = w_lat + (size_t)j * 4;
  float v = b_lat[j] + q[0] * w[0] + q[1] * w[1] + q[2] * w[2] + q[3] * w[3];
  lat[idx] = (_Float16)fmaxf(v, 0.f);
}

// ---------------------------------------------------------------------------
// Kernel 4: h = relu(latent @ w_fc^T + b_fc) via v_wmma_f32_16x16x32_f16.
// One 16x16 C tile per wave, 8 waves/block, K=224 -> 7 WMMA steps. h in fp16.
// ---------------------------------------------------------------------------
__global__ __launch_bounds__(256) void k_gemm_wmma(const _Float16* __restrict__ lat,
                                                   const _Float16* __restrict__ wfc16,
                                                   const float* __restrict__ b_fc,
                                                   _Float16* __restrict__ h, int B) {
  const int NT = 25;                       // 400 / 16 tiles in N
  int lane = threadIdx.x & 31;
  int wave = threadIdx.x >> 5;
  int tile = blockIdx.x * 8 + wave;
  int tm = tile / NT;
  int tn = tile - tm * NT;
  if (tm >= (B >> 4)) return;              // wave-uniform
  int hi   = lane >> 4;                    // lane half selects K sub-range
  int mrow = tm * 16 + (lane & 15);
  int ncol = tn * 16 + (lane & 15);
  const _Float16* arow = lat   + (size_t)mrow * 224;
  const _Float16* brow = wfc16 + (size_t)ncol * 224;   // B column-major
  v8f acc = {};
#pragma unroll
  for (int k0 = 0; k0 < 224; k0 += 32) {
    // A 16x32 f16 layout: lane<16 -> K[k0..k0+7],[k0+16..k0+23];
    //                     lane>=16 -> K[k0+8..k0+15],[k0+24..k0+31]
    int kA = k0 + hi * 8;
    v8h a0 = *(const v8h*)(arow + kA);        // 16B aligned (224*2 = 448)
    v8h a1 = *(const v8h*)(arow + kA + 16);
    v16h a;
#pragma unroll
    for (int i = 0; i < 8; i++) { a[i] = a0[i]; a[i + 8] = a1[i]; }
    // B 32x16 f16: lane<16 -> col n, K[k0..k0+7],[k0+16..k0+23]; lane>=16 -> K[k0+8..k0+15],[k0+24..k0+31]
    int kB = k0 + hi * 8;
    v8h b0 = *(const v8h*)(brow + kB);        // 16B aligned
    v8h b1 = *(const v8h*)(brow + kB + 16);
    v16h b;
#pragma unroll
    for (int i = 0; i < 8; i++) { b[i] = b0[i]; b[i + 8] = b1[i]; }
    acc = __builtin_amdgcn_wmma_f32_16x16x32_f16(
        /*neg_a=*/false, a, /*neg_b=*/false, b,
        /*c_mod=*/(short)0, acc, /*reuse_a=*/false, /*reuse_b=*/false);
  }
  if (ncol < 392) {                         // drop padded N columns
    float bias = b_fc[ncol];
    int mbase = tm * 16 + hi * 8;           // C layout: VGPR j -> M=j (+8 hi half)
#pragma unroll
    for (int j = 0; j < 8; j++) {
      float v = acc[j] + bias;
      h[(size_t)(mbase + j) * 392 + ncol] = (_Float16)fmaxf(v, 0.f);
    }
  }
}

// ---------------------------------------------------------------------------
// Kernel 5: fused deconv1(2x2,s2)+ReLU -> deconv2(2x2,s2) -> sigmoid.
// One block per sample; h tile staged into LDS via async b128 copies
// (ASYNCcnt path) when available; float4 output stores.
// ---------------------------------------------------------------------------
__global__ __launch_bounds__(256) void k_deconv(const _Float16* __restrict__ h,
                                                const float* __restrict__ w_d1,
                                                const float* __restrict__ b_d1,
                                                const float* __restrict__ w_d2,
                                                const float* __restrict__ b_d2,
                                                float* __restrict__ out) {
  __shared__ __align__(16) _Float16 h_s[392];
  __shared__ float w1_s[128];   // (8,4,2,2)
  __shared__ float w2_s[16];    // (4,1,2,2)
  __shared__ float b1_s[4];
  __shared__ float b2_s;
  int s = blockIdx.x;
  int tid = threadIdx.x;
  const _Float16* hp = h + (size_t)s * 392;   // 784B/sample, 16B aligned
#if ASYNC_LDS
  if (tid < 49) {                             // 49 x b128 = 784 bytes
    __builtin_amdgcn_global_load_async_to_lds_b128(
        (as1_v4i_p)(hp + tid * 8),            // v4i* in AS1 (non-const)
        (as3_v4i_p)(&h_s[tid * 8]),           // v4i* in AS3
        0, 0);
  }
#else
  for (int i = tid; i < 392; i += 256) h_s[i] = hp[i];
#endif
  if (tid < 128)       w1_s[tid] = w_d1[tid];
  else if (tid < 144)  w2_s[tid - 128] = w_d2[tid - 128];
  else if (tid < 148)  b1_s[tid - 144] = b_d1[tid - 144];
  else if (tid == 148) b2_s = b_d2[0];
#if ASYNC_LDS
  __builtin_amdgcn_s_wait_asynccnt(0);        // own async copies landed in LDS
#endif
  __syncthreads();                            // all waves' copies visible
  if (tid < 196) {                            // one float4 of output per thread
    int p = tid * 4;                          // 28 % 4 == 0 -> same row
    int r = p / 28, cbase = p - r * 28;
    int R = r >> 1;
    int k2r = (r & 1) * 2;
    float res[4];
#pragma unroll
    for (int i = 0; i < 4; i++) {
      int c = cbase + i;
      int C = c >> 1;
      int hoff = (R >> 1) * 7 + (C >> 1);     // 7x7 source position
      int k1 = (R & 1) * 2 + (C & 1);         // deconv1 kernel tap
      int k2 = k2r + (c & 1);                 // deconv2 kernel tap
      float acc2 = b2_s;
#pragma unroll
      for (int ci = 0; ci < 4; ci++) {
        float h1 = b1_s[ci];
#pragma unroll
        for (int c8 = 0; c8 < 8; c8++)
          h1 += (float)h_s[c8 * 49 + hoff] * w1_s[c8 * 16 + ci * 4 + k1];
        h1 = fmaxf(h1, 0.f);
        acc2 += h1 * w2_s[ci * 4 + k2];
      }
      res[i] = 1.f / (1.f + __expf(-acc2));
    }
    float4 o4 = { res[0], res[1], res[2], res[3] };
    *(float4*)(out + (size_t)s * 784 + p) = o4;   // b128 store
  }
}

// ---------------------------------------------------------------------------
extern "C" void kernel_launch(void* const* d_in, const int* in_sizes, int n_in,
                              void* d_out, int out_size, void* d_ws, size_t ws_size,
                              hipStream_t stream) {
  const float* x          = (const float*)d_in[0];
  const float* var_params = (const float*)d_in[1];
  const float* w_lat      = (const float*)d_in[2];
  const float* b_lat      = (const float*)d_in[3];
  const float* w_fc       = (const float*)d_in[4];
  const float* b_fc       = (const float*)d_in[5];
  const float* w_d1       = (const float*)d_in[6];
  const float* b_d1       = (const float*)d_in[7];
  const float* w_d2       = (const float*)d_in[8];
  const float* b_d2       = (const float*)d_in[9];
  float* out = (float*)d_out;
  int B = in_sizes[0] / 784;               // 16384

  // workspace carve-out (256B-aligned slices)
  char* ws = (char*)d_ws;
  size_t off = 0;
  auto take = [&](size_t bytes) -> char* {
    char* p = ws + off;
    off = (off + bytes + 255) & ~(size_t)255;
    return p;
  };
  float*    qf    = (float*)   take((size_t)B * 4   * sizeof(float));
  _Float16* lat   = (_Float16*)take((size_t)B * 224 * sizeof(_Float16));
  _Float16* wfc16 = (_Float16*)take((size_t)400 * 224 * sizeof(_Float16));
  _Float16* hbuf  = (_Float16*)take((size_t)B * 392 * sizeof(_Float16));
  (void)ws_size; (void)n_in; (void)out_size;

  k_qfeat   <<<(B + 7) / 8, 256, 0, stream>>>(x, var_params, qf, B);
  k_pack_wfc<<<(400 * 224 + 255) / 256, 256, 0, stream>>>(w_fc, wfc16);
  k_latent  <<<(B * 224 + 255) / 256, 256, 0, stream>>>(qf, w_lat, b_lat, lat, B);
  int tiles = (B / 16) * 25;
  k_gemm_wmma<<<(tiles + 7) / 8, 256, 0, stream>>>(lat, wfc16, b_fc, hbuf, B);
  k_deconv  <<<B, 256, 0, stream>>>(hbuf, w_d1, b_d1, w_d2, b_d2, out);
}
